// Encoder_22995254903419
// MI455X (gfx1250) — matrix operand order repaired and checked by reading, not compiled, not run on hardware
//
#include <hip/hip_runtime.h>
#include <cstdint>
#include <cstddef>

#define LAYERS  4
#define D_MODEL 1024
#define HEADS   16
#define HIDDEN  4096
#define BSZ     2
#define SEQ     2048
#define DK      64
#define NTOK    (BSZ * SEQ)      /* 4096 tokens */
#define QKV_W   (3 * D_MODEL)    /* 3072 */

typedef __attribute__((ext_vector_type(16))) __bf16 v16bf;
typedef __attribute__((ext_vector_type(8)))  float  v8f;
typedef __attribute__((ext_vector_type(4)))  unsigned int u32x4;
typedef __attribute__((ext_vector_type(8)))  int i32x8;
typedef __attribute__((ext_vector_type(4)))  int i32x4;

#if defined(__has_builtin)
#if __has_builtin(__builtin_amdgcn_tensor_load_to_lds) && \
    __has_builtin(__builtin_amdgcn_s_wait_tensorcnt)
#define USE_TDM 1
#endif
#endif
#ifndef USE_TDM
#define USE_TDM 0
#endif

struct __align__(16) U128 { unsigned int x, y, z, w; };
struct __align__(16) F128 { float x, y, z, w; };
struct __align__(8)  BF4  { __bf16 a, b, c, d; };

union FragAB {
  v16bf v;
  U128  u[2];
};

__device__ __forceinline__ v8f wmma_bf16(const FragAB& a, const FragAB& b, v8f c) {
  // D(f32 16x16) = A(bf16 16x32) * B(bf16 32x16) + C
  return __builtin_amdgcn_wmma_f32_16x16x32_bf16(false, a.v, false, b.v,
                                                 (short)0, c, false, false);
}

#if USE_TDM
// ---------------------------------------------------------------------------
// TDM: DMA a 2D tile (tile_d1 rows x tile_d0 elems of 2B) from global memory
// (row stride = stride_elems) into LDS at lds_addr, inserting LDS row padding
// (pad_amount encoding: 3 -> 4 DWORDs = 16 B) every 2^(pad_interval+1) DWORDs.
// D# layout per CDNA5 ISA. Issued by one wave;
// completion tracked with TENSORcnt.
// 6-arg builtin form (clang-23 / amdgpu-toolchain lane).
// ---------------------------------------------------------------------------
__device__ __forceinline__ void tdm_load_2d_bf16(
    unsigned lds_addr, const void* gptr,
    unsigned tile_d0, unsigned tile_d1,
    unsigned stride_elems, unsigned tensor_d1,
    unsigned pad_interval, unsigned pad_amount)
{
  const unsigned long long ga = (unsigned long long)gptr;
  const unsigned dim0 = stride_elems;           // logical dim0 length
  const u32x4 g0 = {
      1u,                                        // count=1, user descriptor
      lds_addr,                                  // lds_addr [63:32]
      (unsigned)ga,                              // global_addr [95:64]
      (unsigned)(ga >> 32) | (2u << 30)          // global_addr hi | type=2
  };
  const i32x8 g1 = {
      (int)((1u << 16) | (1u << 20) |            // data_size=2B, pad_enable
            (pad_interval << 22) | (pad_amount << 25)),
      (int)((dim0 & 0xFFFFu) << 16),             // tensor_dim0[15:0]
      (int)(((dim0 >> 16) & 0xFFFFu) | ((tensor_d1 & 0xFFFFu) << 16)),
      (int)(((tensor_d1 >> 16) & 0xFFFFu) | (tile_d0 << 16)),  // tile_dim0
      (int)(tile_d1 & 0xFFFFu),                  // tile_dim1 (tile_dim2=0)
      (int)stride_elems,                         // tensor_dim0_stride[31:0]
      0,                                         // stride[47:32] | dim1_stride lo
      0
  };
  const i32x4 z4 = {0, 0, 0, 0};
  const i32x8 z8 = {0, 0, 0, 0, 0, 0, 0, 0};
  __builtin_amdgcn_tensor_load_to_lds(g0, g1, z4, z4, z8, 0);
}
#endif

// ---------------------------------------------------------------------------
// Elementwise helpers
// ---------------------------------------------------------------------------
__global__ void cvt_f32_bf16_v4(const F128* __restrict__ in,
                                BF4* __restrict__ out, size_t n4) {
  size_t i = (size_t)blockIdx.x * blockDim.x + threadIdx.x;
  const size_t stride = (size_t)gridDim.x * blockDim.x;
  for (; i < n4; i += stride) {
    const F128 f = in[i];
    BF4 o;
    o.a = (__bf16)f.x; o.b = (__bf16)f.y; o.c = (__bf16)f.z; o.d = (__bf16)f.w;
    out[i] = o;
  }
}

__global__ void copy_f32_v4(const F128* __restrict__ in,
                            F128* __restrict__ out, size_t n4) {
  size_t i = (size_t)blockIdx.x * blockDim.x + threadIdx.x;
  const size_t stride = (size_t)gridDim.x * blockDim.x;
  for (; i < n4; i += stride) out[i] = in[i];
}

// ---------------------------------------------------------------------------
// Tiled bf16 WMMA GEMM:  out = A(MxK, bf16) @ Bw(KxN, bf16) + bias [+resid]
//   RESID: add f32 residual and store f32 result to outF (residual stream)
//   GELU : exact GELU on the biased result
// Always stores bf16 result to outB. Double-buffered LDS; A tile staged by
// the Tensor Data Mover (TENSORcnt), B tile transposed manually (TDM cannot
// transpose). One barrier per K-step.
// ---------------------------------------------------------------------------
#define BM 128
#define BN 128
#define BK 32

template <bool RESID, bool GELU>
__global__ __launch_bounds__(256) void gemm_bf16_wmma(
    const __bf16* __restrict__ A, const __bf16* __restrict__ Bw,
    const float* __restrict__ bias, const float* __restrict__ resid,
    float* __restrict__ outF, __bf16* __restrict__ outB,
    int N, int K)
{
  __shared__ __align__(16) __bf16 At[2][BM][BK + 8];  // 80 B row stride
  __shared__ __align__(16) __bf16 Bt[2][BN][BK + 8];  // transposed [n][k]

  const int tid  = threadIdx.x;
  const int lane = tid & 31;
  const int w    = tid >> 5;
  const int hi   = lane >> 4;
  const int l15  = lane & 15;
  const int m0   = blockIdx.y * BM;
  const int n0   = blockIdx.x * BN;
  const int wm   = (w >> 2) * 64;
  const int wn   = (w & 3) * 32;

  const int sr = tid >> 1;            // A row 0..127 (manual path)
  const int sc = (tid & 1) * 16;      // A col chunk
  const int sk = tid >> 3;            // B k 0..31
  const int sn = (tid & 7) * 16;      // B n chunk

  auto stageA = [&](int buf, int k0) {
#if USE_TDM
    if (w == 0) {
      // 128 rows x 32 elems (64 B) per row; LDS pad 16 B per row:
      // pad_interval=3 (16 DWORDs), pad_amount=3 (4 DWORDs).
      tdm_load_2d_bf16((unsigned)(uintptr_t)&At[buf][0][0],
                       A + (size_t)m0 * K + k0,
                       BK, BM, (unsigned)K, (unsigned)NTOK, 3u, 3u);
    }
#else
    const __bf16* ga = A + (size_t)(m0 + sr) * K + k0 + sc;
    *(U128*)(&At[buf][sr][sc])     = *(const U128*)(ga);
    *(U128*)(&At[buf][sr][sc + 8]) = *(const U128*)(ga + 8);
#endif
  };
  auto stageB = [&](int buf, int k0) {
    const __bf16* gb = Bw + (size_t)(k0 + sk) * N + n0 + sn;
#pragma unroll
    for (int e = 0; e < 16; ++e) Bt[buf][sn + e][sk] = gb[e];
  };

  const v8f vzero = {0.f, 0.f, 0.f, 0.f, 0.f, 0.f, 0.f, 0.f};
  v8f acc[4][2];
#pragma unroll
  for (int mi = 0; mi < 4; ++mi)
#pragma unroll
    for (int ni = 0; ni < 2; ++ni) acc[mi][ni] = vzero;

  stageB(0, 0);
  stageA(0, 0);
  int cur = 0;
  for (int k0 = 0; k0 < K; k0 += BK) {
#if USE_TDM
    if (w == 0) __builtin_amdgcn_s_wait_tensorcnt(0);  // A tile DMA complete
#endif
    __syncthreads();                       // buf[cur] staged; buf[cur^1] free
    if (k0 + BK < K) {                     // overlap next staging with WMMA
      stageB(cur ^ 1, k0 + BK);
      stageA(cur ^ 1, k0 + BK);
    }

    FragAB af[4], bfr[2];
#pragma unroll
    for (int mi = 0; mi < 4; ++mi) {
      const int row = wm + mi * 16 + l15;
      const int cb  = hi ? 8 : 0;
      af[mi].u[0] = *(const U128*)(&At[cur][row][cb]);
      af[mi].u[1] = *(const U128*)(&At[cur][row][cb + 16]);
    }
#pragma unroll
    for (int ni = 0; ni < 2; ++ni) {
      const int nn = wn + ni * 16 + l15;
      const int cb = hi ? 8 : 0;
      bfr[ni].u[0] = *(const U128*)(&Bt[cur][nn][cb]);
      bfr[ni].u[1] = *(const U128*)(&Bt[cur][nn][cb + 16]);
    }
#pragma unroll
    for (int mi = 0; mi < 4; ++mi)
#pragma unroll
      for (int ni = 0; ni < 2; ++ni)
        acc[mi][ni] = wmma_bf16(af[mi], bfr[ni], acc[mi][ni]);
    cur ^= 1;
  }

  // Branch-free epilogue.
#pragma unroll
  for (int mi = 0; mi < 4; ++mi) {
    const int rowb = m0 + wm + mi * 16 + hi * 8;
#pragma unroll
    for (int ni = 0; ni < 2; ++ni) {
      const int col = n0 + wn + ni * 16 + l15;
      const float bv = bias[col];
      float v[8];
#pragma unroll
      for (int r = 0; r < 8; ++r) v[r] = acc[mi][ni][r] + bv;
      if constexpr (RESID) {
        float rv[8];
#pragma unroll
        for (int r = 0; r < 8; ++r)
          rv[r] = resid[(size_t)(rowb + r) * N + col];
#pragma unroll
        for (int r = 0; r < 8; ++r) v[r] += rv[r];
      }
      if constexpr (GELU) {
#pragma unroll
        for (int r = 0; r < 8; ++r)
          v[r] = 0.5f * v[r] * (1.0f + erff(v[r] * 0.70710678118f));
      }
#pragma unroll
      for (int r = 0; r < 8; ++r) {
        const size_t idx = (size_t)(rowb + r) * N + col;
        if constexpr (RESID) outF[idx] = v[r];
        outB[idx] = (__bf16)v[r];
      }
    }
  }
}

// ---------------------------------------------------------------------------
// Flash attention, bf16 WMMA, f32 accumulation, double-buffered K/V staging.
// K tile staged by TDM ([key][dk], contiguous rows); V staged manually
// transposed to [dk][key] so P@V B-fragments are contiguous LDS b128 reads.
// ---------------------------------------------------------------------------
__global__ __launch_bounds__(256) void attn_flash_wmma(
    const __bf16* __restrict__ qkv, __bf16* __restrict__ ctx)
{
  __shared__ __align__(16) __bf16 kt[2][64][72];    // [key][dk]
  __shared__ __align__(16) __bf16 vt[2][64][72];    // [dk][key] (transposed)
  __shared__ __align__(16) __bf16 pt[8][16][72];    // per-wave P slab

  const int tid  = threadIdx.x;
  const int lane = tid & 31;
  const int w    = tid >> 5;
  const int hi   = lane >> 4;
  const int l15  = lane & 15;

  const int b  = blockIdx.z;
  const int h  = blockIdx.y;
  const int q0 = blockIdx.x * 128;

  const __bf16* Qp = qkv + (size_t)b * SEQ * QKV_W + h * DK;
  const __bf16* Kp = Qp + D_MODEL;
  const __bf16* Vp = Qp + 2 * D_MODEL;

  const int skr = tid >> 2;          // key row 0..63
  const int sdc = (tid & 3) * 16;    // dk chunk

  auto stageK = [&](int buf, int kv) {
#if USE_TDM
    if (w == 0) {
      // 64 rows x 64 elems (128 B); LDS pad 16 B per row:
      // pad_interval=4 (32 DWORDs), pad_amount=3 (4 DWORDs).
      tdm_load_2d_bf16((unsigned)(uintptr_t)&kt[buf][0][0],
                       Kp + (size_t)kv * QKV_W,
                       DK, 64, (unsigned)QKV_W, (unsigned)NTOK, 4u, 3u);
    }
#else
    const __bf16* kg = Kp + (size_t)(kv + skr) * QKV_W + sdc;
    *(U128*)(&kt[buf][skr][sdc])     = *(const U128*)(kg);
    *(U128*)(&kt[buf][skr][sdc + 8]) = *(const U128*)(kg + 8);
#endif
  };
  auto stageV = [&](int buf, int kv) {
    const __bf16* vg = Vp + (size_t)(kv + skr) * QKV_W + sdc;
#pragma unroll
    for (int e = 0; e < 16; ++e) vt[buf][sdc + e][skr] = vg[e];
  };

  // Persistent Q fragments (A-matrix layout) for this wave's 16 rows.
  FragAB qf[2];
  {
    const int qrow = q0 + w * 16 + l15;
    const __bf16* qr = Qp + (size_t)qrow * QKV_W;
#pragma unroll
    for (int kk = 0; kk < 2; ++kk) {
      const int cb = kk * 32 + (hi ? 8 : 0);
      qf[kk].u[0] = *(const U128*)(qr + cb);
      qf[kk].u[1] = *(const U128*)(qr + cb + 16);
    }
  }

  const v8f vzero = {0.f, 0.f, 0.f, 0.f, 0.f, 0.f, 0.f, 0.f};
  v8f ov[4];
#pragma unroll
  for (int t = 0; t < 4; ++t) ov[t] = vzero;

  float mrow[8], lrow[8];
#pragma unroll
  for (int r = 0; r < 8; ++r) { mrow[r] = -1e30f; lrow[r] = 0.f; }

  const float scale = 0.125f;  // 1/sqrt(64)

  stageV(0, 0);
  stageK(0, 0);
  int cur = 0;
  for (int kv = 0; kv < SEQ; kv += 64) {
#if USE_TDM
    if (w == 0) __builtin_amdgcn_s_wait_tensorcnt(0);  // K tile DMA complete
#endif
    __syncthreads();                         // buf[cur] ready; buf[cur^1] free
    if (kv + 64 < SEQ) {
      stageV(cur ^ 1, kv + 64);
      stageK(cur ^ 1, kv + 64);
    }

    // S = Q K^T : 4 key tiles of 16, K-dim = 64 (2 WMMA steps each).
    v8f st[4];
#pragma unroll
    for (int ni = 0; ni < 4; ++ni) {
      st[ni] = vzero;
#pragma unroll
      for (int kk = 0; kk < 2; ++kk) {
        FragAB kf;
        const int krow = ni * 16 + l15;
        const int cb = kk * 32 + (hi ? 8 : 0);
        kf.u[0] = *(const U128*)(&kt[cur][krow][cb]);
        kf.u[1] = *(const U128*)(&kt[cur][krow][cb + 16]);
        st[ni] = wmma_bf16(qf[kk], kf, st[ni]);
      }
    }

    // Online softmax: row stats per 16-lane half (matches C layout).
    float mnew[8];
#pragma unroll
    for (int r = 0; r < 8; ++r) mnew[r] = mrow[r];
#pragma unroll
    for (int ni = 0; ni < 4; ++ni)
#pragma unroll
      for (int r = 0; r < 8; ++r) mnew[r] = fmaxf(mnew[r], st[ni][r] * scale);
#pragma unroll
    for (int r = 0; r < 8; ++r) {
#pragma unroll
      for (int d = 1; d < 16; d <<= 1)
        mnew[r] = fmaxf(mnew[r], __shfl_xor(mnew[r], d, 32));
    }
    float alpha[8], rsum[8];
#pragma unroll
    for (int r = 0; r < 8; ++r) {
      alpha[r] = __expf(mrow[r] - mnew[r]);
      mrow[r]  = mnew[r];
      rsum[r]  = 0.f;
    }
#pragma unroll
    for (int ni = 0; ni < 4; ++ni) {
#pragma unroll
      for (int r = 0; r < 8; ++r) {
        const float p = __expf(st[ni][r] * scale - mnew[r]);
        rsum[r] += p;
        pt[w][r + hi * 8][ni * 16 + l15] = (__bf16)p;
      }
    }
#pragma unroll
    for (int r = 0; r < 8; ++r) {
#pragma unroll
      for (int d = 1; d < 16; d <<= 1) rsum[r] += __shfl_xor(rsum[r], d, 32);
      lrow[r] = lrow[r] * alpha[r] + rsum[r];
    }
#pragma unroll
    for (int t = 0; t < 4; ++t)
#pragma unroll
      for (int r = 0; r < 8; ++r) ov[t][r] *= alpha[r];

    // P fragments (intra-wave LDS round trip; DS ops are in-order per wave).
    FragAB pf[2];
#pragma unroll
    for (int kk = 0; kk < 2; ++kk) {
      const int cb = kk * 32 + (hi ? 8 : 0);
      pf[kk].u[0] = *(const U128*)(&pt[w][l15][cb]);
      pf[kk].u[1] = *(const U128*)(&pt[w][l15][cb + 16]);
    }
    // O += P V : 4 dk tiles, K-dim = 64 keys (2 WMMA steps each).
#pragma unroll
    for (int t = 0; t < 4; ++t) {
#pragma unroll
      for (int kk = 0; kk < 2; ++kk) {
        FragAB vf;
        const int dkr = t * 16 + l15;
        const int cb  = kk * 32 + (hi ? 8 : 0);
        vf.u[0] = *(const U128*)(&vt[cur][dkr][cb]);
        vf.u[1] = *(const U128*)(&vt[cur][dkr][cb + 16]);
        ov[t] = wmma_bf16(pf[kk], vf, ov[t]);
      }
    }
    cur ^= 1;
  }

  // Normalize and emit bf16 context (feeds the out-projection GEMM).
#pragma unroll
  for (int t = 0; t < 4; ++t) {
    const int col = h * DK + t * 16 + l15;
#pragma unroll
    for (int r = 0; r < 8; ++r) {
      const int row = q0 + w * 16 + r + hi * 8;
      const float o = ov[t][r] / lrow[r];
      ctx[(size_t)(b * SEQ + row) * D_MODEL + col] = (__bf16)o;
    }
  }
}

// ---------------------------------------------------------------------------
// Host-side orchestration
// ---------------------------------------------------------------------------
extern "C" void kernel_launch(void* const* d_in, const int* in_sizes, int n_in,
                              void* d_out, int out_size, void* d_ws, size_t ws_size,
                              hipStream_t stream) {
  (void)in_sizes; (void)n_in; (void)out_size; (void)ws_size;

  const float* x    = (const float*)d_in[0];
  const float* Wqkv = (const float*)d_in[1];
  const float* bqkv = (const float*)d_in[2];
  const float* Wout = (const float*)d_in[3];
  const float* bout = (const float*)d_in[4];
  const float* W1   = (const float*)d_in[5];
  const float* b1   = (const float*)d_in[6];
  const float* W2   = (const float*)d_in[7];
  const float* b2   = (const float*)d_in[8];

  char* ws = (char*)d_ws;
  size_t off = 0;
  auto carve = [&](size_t bytes) -> void* {
    void* p = ws + off;
    off = (off + bytes + 255) & ~(size_t)255;
    return p;
  };
  float*  hA   = (float*)  carve((size_t)NTOK * D_MODEL * 4);   // f32 residual
  __bf16* hB   = (__bf16*) carve((size_t)NTOK * D_MODEL * 2);   // bf16 copy
  __bf16* qkvB = (__bf16*) carve((size_t)NTOK * QKV_W * 2);
  __bf16* ctxB = (__bf16*) carve((size_t)NTOK * D_MODEL * 2);
  __bf16* ffB  = (__bf16*) carve((size_t)NTOK * HIDDEN * 2);
  __bf16* wbuf = (__bf16*) carve((size_t)D_MODEL * HIDDEN * 2); // weight staging

  const dim3 blk(256);
  const dim3 gQKV(QKV_W / BN,   NTOK / BM);  // 24 x 32
  const dim3 gOUT(D_MODEL / BN, NTOK / BM);  //  8 x 32
  const dim3 gFF1(HIDDEN / BN,  NTOK / BM);  // 32 x 32
  const dim3 gATT(SEQ / 128, HEADS, BSZ);    // 16 x 16 x 2

  auto cvt = [&](const float* src, __bf16* dst, size_t n) {
    cvt_f32_bf16_v4<<<1024, 256, 0, stream>>>((const F128*)src, (BF4*)dst, n / 4);
  };

  copy_f32_v4<<<1024, 256, 0, stream>>>((const F128*)x, (F128*)hA,
                                        (size_t)NTOK * D_MODEL / 4);
  cvt(x, hB, (size_t)NTOK * D_MODEL);

  for (int l = 0; l < LAYERS; ++l) {
    // QKV projection -> bf16
    cvt(Wqkv + (size_t)l * D_MODEL * QKV_W, wbuf, (size_t)D_MODEL * QKV_W);
    gemm_bf16_wmma<false, false><<<gQKV, blk, 0, stream>>>(
        hB, wbuf, bqkv + (size_t)l * QKV_W, nullptr, nullptr, qkvB,
        QKV_W, D_MODEL);
    // Flash attention -> bf16 context
    attn_flash_wmma<<<gATT, blk, 0, stream>>>(qkvB, ctxB);

    // Output projection + residual: updates hA (f32) and hB (bf16)
    cvt(Wout + (size_t)l * D_MODEL * D_MODEL, wbuf, (size_t)D_MODEL * D_MODEL);
    gemm_bf16_wmma<true, false><<<gOUT, blk, 0, stream>>>(
        ctxB, wbuf, bout + (size_t)l * D_MODEL, hA, hA, hB,
        D_MODEL, D_MODEL);
    // FFN up + exact GELU -> bf16 hidden
    cvt(W1 + (size_t)l * D_MODEL * HIDDEN, wbuf, (size_t)D_MODEL * HIDDEN);
    gemm_bf16_wmma<false, true><<<gFF1, blk, 0, stream>>>(
        hB, wbuf, b1 + (size_t)l * HIDDEN, nullptr, nullptr, ffB,
        HIDDEN, D_MODEL);
    // FFN down + residual
    cvt(W2 + (size_t)l * HIDDEN * D_MODEL, wbuf, (size_t)HIDDEN * D_MODEL);
    gemm_bf16_wmma<true, false><<<gOUT, blk, 0, stream>>>(
        ffB, wbuf, b2 + (size_t)l * D_MODEL, hA, hA, hB,
        D_MODEL, HIDDEN);
  }

  copy_f32_v4<<<1024, 256, 0, stream>>>((const F128*)hA, (F128*)d_out,
                                        (size_t)NTOK * D_MODEL / 4);
}
